// GatNet_90623809946227
// MI455X (gfx1250) — compile-verified, matrix operand order checked by reading, not run on hardware
//
#include <hip/hip_runtime.h>
#include <hip/hip_bf16.h>

typedef __attribute__((ext_vector_type(16))) __bf16 v16bf;
typedef __attribute__((ext_vector_type(8)))  float  v8f;
typedef unsigned short ushort_t;
typedef unsigned int   uint_t;

#define NNODE 100000
#define NEDGE 3200000
#define NB    32
#define LSEQ  1000
#define DIM   128
#define FDIM  384

// ---------- helpers ----------
__device__ __forceinline__ ushort_t f2bf(float f) {
    uint_t u = __float_as_uint(f);
    u = u + 0x7FFFu + ((u >> 16) & 1u);
    return (ushort_t)(u >> 16);
}
__device__ __forceinline__ float bf2f(ushort_t h) {
    return __uint_as_float(((uint_t)h) << 16);
}
// order-preserving float<->uint encoding for atomicMax on floats
__device__ __forceinline__ uint_t fenc(float f) {
    uint_t u = __float_as_uint(f);
    return (u & 0x80000000u) ? ~u : (u | 0x80000000u);
}
__device__ __forceinline__ float fdec(uint_t u) {
    uint_t v = (u & 0x80000000u) ? (u & 0x7FFFFFFFu) : ~u;
    return __uint_as_float(v);
}
#define ENC_NEG_INF 0x007FFFFFu   // fenc(-inf)

// ---------- generic fill / cast ----------
__global__ void fill_u32(uint_t* p, uint_t v, long long n) {
    long long i = (long long)blockIdx.x * blockDim.x + threadIdx.x;
    if (i < n) p[i] = v;
}
__global__ void cast_f32_bf16(const float* in, ushort_t* out, long long n) {
    long long i = (long long)blockIdx.x * blockDim.x + threadIdx.x;
    if (i < n) out[i] = f2bf(in[i]);
}
// W (K x Nc) row-major -> Wt (Nc x K) bf16 row-major
__global__ void cast_transpose_bf16(const float* W, ushort_t* Wt, int K, int Nc) {
    long long i = (long long)blockIdx.x * blockDim.x + threadIdx.x;
    if (i >= (long long)K * Nc) return;
    int k = (int)(i / Nc), n = (int)(i % Nc);
    Wt[(long long)n * K + k] = f2bf(W[i]);
}
// pad_dmap (B,1,L,D) -> xb (B,D,L) bf16
__global__ void cast_pad_transpose(const float* pad, ushort_t* xb) {
    long long i = (long long)blockIdx.x * blockDim.x + threadIdx.x;
    if (i >= (long long)NB * DIM * LSEQ) return;
    int l = (int)(i % LSEQ);
    int d = (int)((i / LSEQ) % DIM);
    int b = (int)(i / ((long long)DIM * LSEQ));
    xb[i] = f2bf(pad[(long long)b * LSEQ * DIM + (long long)l * DIM + d]);
}

// ---------- WMMA GEMM with LDS-staged weight panel ----------
// C(MxNc,f32) = A(MxK,bf16 rowmajor) * Bt(NcxK,bf16 rowmajor)^T
// Block = 256 threads = 8 waves. Block covers 8 consecutive M-tiles x one
// 64-column N-group. The 64xK bf16 weight panel (<=48KB) is staged in LDS
// once and re-read by all 8 waves via DS; each wave holds 4 accumulators
// (16x64 output) so the A fragment is loaded once per K-step.
__global__ void wmma_gemm_lds(const ushort_t* A, const ushort_t* Bt, float* C,
                              int M, int K, int Nc) {
    __shared__ __align__(16) ushort_t sB[64 * 384];
    const int ngroups = Nc >> 6;
    const int mtiles  = (M + 15) >> 4;
    const int ng = blockIdx.x % ngroups;
    const int mb = blockIdx.x / ngroups;
    const int wv   = threadIdx.x >> 5;
    const int lane = threadIdx.x & 31;

    // cooperative panel load (global -> LDS)
    {
        const uint4* gp = (const uint4*)(Bt + (size_t)ng * 64 * K);
        uint4* sp = (uint4*)sB;
        const int cnt = (64 * K) >> 3;          // uint4 elements
        for (int t = threadIdx.x; t < cnt; t += blockDim.x) sp[t] = gp[t];
    }
    __syncthreads();

    int mtile = mb * 8 + wv;
    const bool active = (mtile < mtiles);
    if (!active) mtile = mtiles - 1;

    int rowA = mtile * 16 + (lane & 15); if (rowA >= M) rowA = M - 1;
    const int koff = (lane >> 4) * 8;
    const uint_t* ap = (const uint_t*)(A + (size_t)rowA * K);

    const uint_t* lp[4];
#pragma unroll
    for (int s = 0; s < 4; ++s)
        lp[s] = (const uint_t*)(sB + (size_t)(s * 16 + (lane & 15)) * K);

    v8f acc[4];
#pragma unroll
    for (int s = 0; s < 4; ++s)
#pragma unroll
        for (int r = 0; r < 8; ++r) acc[s][r] = 0.0f;

    for (int k0 = 0; k0 < K; k0 += 32) {
        const int kb = (k0 + koff) >> 1;        // uint index (2 bf16/uint)
        union { v16bf v; uint_t u[8]; } a;
#pragma unroll
        for (int j = 0; j < 4; ++j) { a.u[j] = ap[kb + j]; a.u[4 + j] = ap[kb + 8 + j]; }
#pragma unroll
        for (int s = 0; s < 4; ++s) {
            union { v16bf v; uint_t u[8]; } b;
#pragma unroll
            for (int j = 0; j < 4; ++j) { b.u[j] = lp[s][kb + j]; b.u[4 + j] = lp[s][kb + 8 + j]; }
            acc[s] = __builtin_amdgcn_wmma_f32_16x16x32_bf16(false, a.v, false, b.v,
                                                             (short)0, acc[s], false, false);
        }
    }
    const int rb = (lane >> 4) * 8;
#pragma unroll
    for (int s = 0; s < 4; ++s) {
        const int col = ng * 64 + s * 16 + (lane & 15);
#pragma unroll
        for (int r = 0; r < 8; ++r) {
            const int row = mtile * 16 + rb + r;
            if (active && row < M) C[(long long)row * Nc + col] = acc[s][r];
        }
    }
}

// ---------- WMMA conv1d: out(b,o,t) = sum_{i,k} w(o,i,k) x(b,i,t+k), K=384 ----------
// Each wave: 16 out-channels x 64 time steps; weight fragment loaded once per
// K-step and reused for 4 WMMAs against gathered patch fragments.
__global__ void wmma_conv(const ushort_t* x, const ushort_t* w, const float* bias,
                          ushort_t* out, int L, int Lout) {
    const int flat = blockIdx.x * blockDim.x + threadIdx.x;
    const int wid  = flat >> 5;
    const int lane = threadIdx.x & 31;
    const int tgroups = (Lout + 63) >> 6;
    if (wid >= NB * 8 * tgroups) return;
    const int b   = wid / (8 * tgroups);
    const int rem = wid % (8 * tgroups);
    const int to  = rem / tgroups;
    const int tg  = rem % tgroups;
    const int tbase = tg * 64;

    const int orow = to * 16 + (lane & 15);
    const int koff = (lane >> 4) * 8;

    const uint_t*   ap = (const uint_t*)(w + (long long)orow * 384);
    const ushort_t* xb = x + (long long)b * DIM * L;

    v8f acc[4];
#pragma unroll
    for (int s = 0; s < 4; ++s)
#pragma unroll
        for (int r = 0; r < 8; ++r) acc[s][r] = 0.0f;

    for (int k0 = 0; k0 < 384; k0 += 32) {
        const int kb = (k0 + koff) >> 1;
        union { v16bf v; uint_t u[8]; } a;
#pragma unroll
        for (int j = 0; j < 4; ++j) { a.u[j] = ap[kb + j]; a.u[4 + j] = ap[kb + 8 + j]; }
#pragma unroll
        for (int s = 0; s < 4; ++s) {
            const int tcol = tbase + s * 16 + (lane & 15);
            union { v16bf v; ushort_t sh[16]; } bfr;
#pragma unroll
            for (int j = 0; j < 16; ++j) {
                int kk = k0 + ((j < 8) ? (koff + j) : (16 + koff + (j - 8)));
                int ic = kk / 3;
                int kc = kk - 3 * ic;
                int t  = tcol + kc; if (t > L - 1) t = L - 1;
                bfr.sh[j] = xb[(long long)ic * L + t];
            }
            acc[s] = __builtin_amdgcn_wmma_f32_16x16x32_bf16(false, a.v, false, bfr.v,
                                                             (short)0, acc[s], false, false);
        }
    }
    const int rb = (lane >> 4) * 8;
#pragma unroll
    for (int s = 0; s < 4; ++s) {
        const int t = tbase + s * 16 + (lane & 15);
        if (t < Lout) {
#pragma unroll
            for (int r = 0; r < 8; ++r) {
                const int o = to * 16 + rb + r;
                out[((long long)b * DIM + o) * Lout + t] = f2bf(acc[s][r] + bias[o]);
            }
        }
    }
}

// ---------- pooling ----------
__global__ void maxpool3(const ushort_t* in, ushort_t* out, int Lin, int Lout) {
    long long i = (long long)blockIdx.x * blockDim.x + threadIdx.x;
    if (i >= (long long)NB * DIM * Lout) return;
    int t = (int)(i % Lout);
    long long c = i / Lout;
    const ushort_t* p = in + c * Lin + 3 * t;
    float m = bf2f(p[0]);
    m = fmaxf(m, bf2f(p[1]));
    m = fmaxf(m, bf2f(p[2]));
    out[i] = f2bf(m);
}
__global__ void pool_all(const ushort_t* in, float* out, int Lin) {
    int c = blockIdx.x * blockDim.x + threadIdx.x;
    if (c >= NB * DIM) return;
    const ushort_t* p = in + (long long)c * Lin;
    float m = bf2f(p[0]);
    for (int t = 1; t < Lin; ++t) m = fmaxf(m, bf2f(p[t]));
    out[c] = m;
}

// ---------- GAT attention coefficients: el/er[n*H+h] = <h1[n,h,:], al/ar[h,:]> ----------
__global__ void gat_attn_coef(const float* hfeat, const float* al, const float* ar,
                              float* el, float* er, int N, int Hh, int od) {
    int wid  = (blockIdx.x * blockDim.x + threadIdx.x) >> 5;
    int lane = threadIdx.x & 31;
    if (wid >= N * Hh) return;
    int n = wid / Hh, hd = wid % Hh;
    const float* hp  = hfeat + (long long)n * (Hh * od) + (long long)hd * od;
    const float* alp = al + hd * od;
    const float* arp = ar + hd * od;
    float sl = 0.f, sr = 0.f;
    for (int d = lane; d < od; d += 32) { float v = hp[d]; sl += v * alp[d]; sr += v * arp[d]; }
    for (int o = 16; o; o >>= 1) { sl += __shfl_xor(sl, o, 32); sr += __shfl_xor(sr, o, 32); }
    if (lane == 0) { el[wid] = sl; er[wid] = sr; }
}

__device__ __forceinline__ float edge_e(const float* el, const float* er,
                                        int s, int d, int Hh, int h) {
    float x = el[s * Hh + h] + er[d * Hh + h];
    return x > 0.f ? x : 0.2f * x;            // leaky_relu(0.2)
}

// pass 1: segment max over dst
__global__ void gat_edge_max(const int* src, const int* dst, const float* el, const float* er,
                             uint_t* m, int E, int Hh) {
    long long i = (long long)blockIdx.x * blockDim.x + threadIdx.x;
    if (i >= (long long)E * Hh) return;
    int e = (int)(i / Hh), h = (int)(i % Hh);
    int s = src[e], d = dst[e];
    atomicMax(&m[d * Hh + h], fenc(edge_e(el, er, s, d, Hh, h)));
}
// pass 2: sum of exp(e - m)
__global__ void gat_edge_sum(const int* src, const int* dst, const float* el, const float* er,
                             const uint_t* m, float* ssum, int E, int Hh) {
    long long i = (long long)blockIdx.x * blockDim.x + threadIdx.x;
    if (i >= (long long)E * Hh) return;
    int e = (int)(i / Hh), h = (int)(i % Hh);
    int s = src[e], d = dst[e];
    float mv = fdec(m[d * Hh + h]);
    if (!(mv > -3.0e38f)) mv = 0.f;          // isolated -> 0 (matches reference)
    float a = __expf(edge_e(el, er, s, d, Hh, h) - mv);
    atomicAdd(&ssum[d * Hh + h], a);
}
// pass 3: rst[dst] += alpha * h[src]  (one wave per edge, lanes cover features)
__global__ void gat_edge_aggr(const int* src, const int* dst, const float* el, const float* er,
                              const uint_t* m, const float* ssum, const float* hfeat,
                              float* rst, int E, int Hh, int od) {
    int wid  = (blockIdx.x * blockDim.x + threadIdx.x) >> 5;
    int lane = threadIdx.x & 31;
    if (wid >= E) return;
    int s = src[wid], d = dst[wid];
    const float* hs = hfeat + (long long)s * (Hh * od);
    float*       rd = rst   + (long long)d * (Hh * od);
    __builtin_prefetch((const void*)(hs + lane * 4), 0, 0);   // warm gather row (L2-resident)
    for (int hh = 0; hh < Hh; ++hh) {
        float mv = fdec(m[d * Hh + hh]);
        if (!(mv > -3.0e38f)) mv = 0.f;
        float alpha = __expf(edge_e(el, er, s, d, Hh, hh) - mv) / ssum[d * Hh + hh];
        for (int f = lane; f < od; f += 32)
            atomicAdd(&rd[hh * od + f], alpha * hs[hh * od + f]);
    }
}

// g_bf16 = bf16(relu(rst + bias))
__global__ void bias_relu_bf16(const float* rst, const float* bias, ushort_t* out, long long n) {
    long long i = (long long)blockIdx.x * blockDim.x + threadIdx.x;
    if (i >= n) return;
    float v = rst[i] + bias[i % FDIM];
    out[i] = f2bf(fmaxf(v, 0.f));
}

// graph max pooling of relu(rst + b2)
__global__ void graph_pool_max(const float* rst, const float* bias, const int* gid,
                               uint_t* gpool, int N) {
    long long i = (long long)blockIdx.x * blockDim.x + threadIdx.x;
    if (i >= (long long)N * FDIM) return;
    int f = (int)(i % FDIM), n = (int)(i / FDIM);
    float v = fmaxf(rst[i] + bias[f], 0.f);
    atomicMax(&gpool[gid[n] * FDIM + f], fenc(v));
}
__global__ void decode_u32(const uint_t* in, float* out, int n) {
    int i = blockIdx.x * blockDim.x + threadIdx.x;
    if (i < n) out[i] = fdec(in[i]);
}

// ---------- small scalar GEMM for head ----------
__global__ void sgemm_small(const float* A, const float* W, const float* bias, float* out,
                            int M, int K, int Nc, int relu) {
    int i = blockIdx.x * blockDim.x + threadIdx.x;
    if (i >= M * Nc) return;
    int mr = i / Nc, n = i % Nc;
    float s = bias ? bias[n] : 0.f;
    const float* a = A + (long long)mr * K;
    for (int k = 0; k < K; ++k) s += a[k] * W[(long long)k * Nc + n];
    if (relu) s = fmaxf(s, 0.f);
    out[i] = s;
}

__global__ void fuse_kernel(const float* g1, const float* seq1, const float* w1, float* gc, int n) {
    int i = blockIdx.x * blockDim.x + threadIdx.x;
    if (i >= n) return;
    float wv = 1.f / (1.f + __expf(-w1[0]));
    gc[i] = (1.f - wv) * g1[i] + wv * seq1[i];
}
__global__ void softmax2(const float* o, float* out) {
    int r = blockIdx.x * blockDim.x + threadIdx.x;
    if (r >= NB) return;
    float a = o[2 * r], b = o[2 * r + 1];
    float mx = fmaxf(a, b);
    float ea = __expf(a - mx), eb = __expf(b - mx);
    float s = ea + eb;
    out[2 * r] = ea / s;
    out[2 * r + 1] = eb / s;
}

// ---------- host ----------
static inline int nblk(long long n, int t) { return (int)((n + t - 1) / t); }

extern "C" void kernel_launch(void* const* d_in, const int* in_sizes, int n_in,
                              void* d_out, int out_size, void* d_ws, size_t ws_size,
                              hipStream_t stream) {
    const float* node_feat = (const float*)d_in[0];
    const int*   src       = (const int*)d_in[1];
    const int*   dst       = (const int*)d_in[2];
    const int*   gids      = (const int*)d_in[3];
    const float* pad_dmap  = (const float*)d_in[4];
    const float* W1  = (const float*)d_in[5];
    const float* al1 = (const float*)d_in[6];
    const float* ar1 = (const float*)d_in[7];
    const float* b1  = (const float*)d_in[8];
    const float* W2  = (const float*)d_in[9];
    const float* al2 = (const float*)d_in[10];
    const float* ar2 = (const float*)d_in[11];
    const float* b2  = (const float*)d_in[12];
    const float* fc_g1_w = (const float*)d_in[13];
    const float* fc_g1_b = (const float*)d_in[14];
    const float* conv1_w = (const float*)d_in[15];
    const float* conv1_b = (const float*)d_in[16];
    const float* conv2_w = (const float*)d_in[17];
    const float* conv2_b = (const float*)d_in[18];
    const float* conv3_w = (const float*)d_in[19];
    const float* conv3_b = (const float*)d_in[20];
    const float* tf_w = (const float*)d_in[21];
    const float* tf_b = (const float*)d_in[22];
    const float* w1s  = (const float*)d_in[23];
    const float* fc1_w = (const float*)d_in[24];
    const float* fc1_b = (const float*)d_in[25];
    const float* fc2_w = (const float*)d_in[26];
    const float* fc2_b = (const float*)d_in[27];
    const float* out_w = (const float*)d_in[28];
    const float* out_b = (const float*)d_in[29];

    char* ws = (char*)d_ws;
    size_t off = 0;
    auto carve = [&](size_t bytes) -> void* {
        void* p = ws + off;
        off = (off + bytes + 255) & ~(size_t)255;
        return p;
    };
    float*    hbuf = (float*)carve((size_t)NNODE * FDIM * 4);    // h1 then h2
    float*    rst  = (float*)carve((size_t)NNODE * FDIM * 4);    // rst1 then rst2, then CNN scratch
    ushort_t* gbf  = (ushort_t*)carve((size_t)NNODE * FDIM * 2); // nf_bf16 then g_bf16
    float*    el   = (float*)carve((size_t)NNODE * 3 * 4);
    float*    er   = (float*)carve((size_t)NNODE * 3 * 4);
    uint_t*   mbuf = (uint_t*)carve((size_t)NNODE * 3 * 4);
    float*    sbuf = (float*)carve((size_t)NNODE * 3 * 4);
    ushort_t* W1t  = (ushort_t*)carve((size_t)FDIM * DIM * 2);
    ushort_t* W2t  = (ushort_t*)carve((size_t)FDIM * FDIM * 2);
    ushort_t* cw1  = (ushort_t*)carve((size_t)DIM * 384 * 2);
    ushort_t* cw2  = (ushort_t*)carve((size_t)DIM * 384 * 2);
    ushort_t* cw3  = (ushort_t*)carve((size_t)DIM * 384 * 2);
    uint_t*   gpool = (uint_t*)carve((size_t)NB * FDIM * 4);
    float*    gpf   = (float*)carve((size_t)NB * FDIM * 4);
    float*    g1    = (float*)carve((size_t)NB * 128 * 4);
    float*    fvec  = (float*)carve((size_t)NB * 128 * 4);
    float*    seq1  = (float*)carve((size_t)NB * 128 * 4);
    float*    gc    = (float*)carve((size_t)NB * 128 * 4);
    float*    t1    = (float*)carve((size_t)NB * 512 * 4);
    float*    t2    = (float*)carve((size_t)NB * 256 * 4);
    float*    o2    = (float*)carve((size_t)NB * 2 * 4);

    // CNN scratch reuses the rst buffer (free after graph_pool_max)
    char* cnn = (char*)rst;
    size_t coff = 0;
    auto ccarve = [&](size_t bytes) -> void* {
        void* p = cnn + coff;
        coff = (coff + bytes + 255) & ~(size_t)255;
        return p;
    };
    ushort_t* xb = (ushort_t*)ccarve((size_t)NB * DIM * LSEQ * 2);
    ushort_t* c1 = (ushort_t*)ccarve((size_t)NB * DIM * 998 * 2);
    ushort_t* p1 = (ushort_t*)ccarve((size_t)NB * DIM * 332 * 2);
    ushort_t* c2 = (ushort_t*)ccarve((size_t)NB * DIM * 330 * 2);
    ushort_t* p2 = (ushort_t*)ccarve((size_t)NB * DIM * 110 * 2);
    ushort_t* c3 = (ushort_t*)ccarve((size_t)NB * DIM * 108 * 2);

    const int T = 256;
    const int mtiles  = (NNODE + 15) / 16;      // 6250
    const int mblocks = (mtiles + 7) / 8;       // 8 M-tiles per block
    const int ngroups = FDIM / 64;              // 6

    // ---- weight & input casts ----
    cast_f32_bf16<<<nblk((long long)NNODE * DIM, T), T, 0, stream>>>(node_feat, gbf, (long long)NNODE * DIM);
    cast_transpose_bf16<<<nblk((long long)DIM * FDIM, T), T, 0, stream>>>(W1, W1t, DIM, FDIM);
    cast_transpose_bf16<<<nblk((long long)FDIM * FDIM, T), T, 0, stream>>>(W2, W2t, FDIM, FDIM);
    cast_f32_bf16<<<nblk((long long)DIM * 384, T), T, 0, stream>>>(conv1_w, cw1, (long long)DIM * 384);
    cast_f32_bf16<<<nblk((long long)DIM * 384, T), T, 0, stream>>>(conv2_w, cw2, (long long)DIM * 384);
    cast_f32_bf16<<<nblk((long long)DIM * 384, T), T, 0, stream>>>(conv3_w, cw3, (long long)DIM * 384);

    // ---- GAT layer 1 (H=3, od=128) ----
    {
        wmma_gemm_lds<<<mblocks * ngroups, T, 0, stream>>>(gbf, W1t, hbuf, NNODE, DIM, FDIM);
        gat_attn_coef<<<nblk((long long)NNODE * 3 * 32, T), T, 0, stream>>>(hbuf, al1, ar1, el, er, NNODE, 3, DIM);
        fill_u32<<<nblk((long long)NNODE * 3, T), T, 0, stream>>>(mbuf, ENC_NEG_INF, (long long)NNODE * 3);
        fill_u32<<<nblk((long long)NNODE * 3, T), T, 0, stream>>>((uint_t*)sbuf, 0u, (long long)NNODE * 3);
        gat_edge_max<<<nblk((long long)NEDGE * 3, T), T, 0, stream>>>(src, dst, el, er, mbuf, NEDGE, 3);
        gat_edge_sum<<<nblk((long long)NEDGE * 3, T), T, 0, stream>>>(src, dst, el, er, mbuf, sbuf, NEDGE, 3);
        fill_u32<<<nblk((long long)NNODE * FDIM, T), T, 0, stream>>>((uint_t*)rst, 0u, (long long)NNODE * FDIM);
        gat_edge_aggr<<<nblk((long long)NEDGE * 32, T), T, 0, stream>>>(src, dst, el, er, mbuf, sbuf, hbuf, rst, NEDGE, 3, DIM);
        bias_relu_bf16<<<nblk((long long)NNODE * FDIM, T), T, 0, stream>>>(rst, b1, gbf, (long long)NNODE * FDIM);
    }
    // ---- GAT layer 2 (H=1, od=384) ----
    {
        wmma_gemm_lds<<<mblocks * ngroups, T, 0, stream>>>(gbf, W2t, hbuf, NNODE, FDIM, FDIM);
        gat_attn_coef<<<nblk((long long)NNODE * 32, T), T, 0, stream>>>(hbuf, al2, ar2, el, er, NNODE, 1, FDIM);
        fill_u32<<<nblk((long long)NNODE, T), T, 0, stream>>>(mbuf, ENC_NEG_INF, (long long)NNODE);
        fill_u32<<<nblk((long long)NNODE, T), T, 0, stream>>>((uint_t*)sbuf, 0u, (long long)NNODE);
        gat_edge_max<<<nblk((long long)NEDGE, T), T, 0, stream>>>(src, dst, el, er, mbuf, NEDGE, 1);
        gat_edge_sum<<<nblk((long long)NEDGE, T), T, 0, stream>>>(src, dst, el, er, mbuf, sbuf, NEDGE, 1);
        fill_u32<<<nblk((long long)NNODE * FDIM, T), T, 0, stream>>>((uint_t*)rst, 0u, (long long)NNODE * FDIM);
        gat_edge_aggr<<<nblk((long long)NEDGE * 32, T), T, 0, stream>>>(src, dst, el, er, mbuf, sbuf, hbuf, rst, NEDGE, 1, FDIM);
    }
    // ---- graph max pooling + fc_g1 ----
    fill_u32<<<nblk((long long)NB * FDIM, T), T, 0, stream>>>(gpool, ENC_NEG_INF, (long long)NB * FDIM);
    graph_pool_max<<<nblk((long long)NNODE * FDIM, T), T, 0, stream>>>(rst, b2, gids, gpool, NNODE);
    decode_u32<<<nblk((long long)NB * FDIM, T), T, 0, stream>>>(gpool, gpf, NB * FDIM);
    sgemm_small<<<nblk((long long)NB * 128, T), T, 0, stream>>>(gpf, fc_g1_w, fc_g1_b, g1, NB, FDIM, 128, 1);

    // ---- text CNN branch (rst region now reusable) ----
    cast_pad_transpose<<<nblk((long long)NB * DIM * LSEQ, T), T, 0, stream>>>(pad_dmap, xb);
    {
        int tg1 = (998 + 63) / 64;
        wmma_conv<<<nblk((long long)NB * 8 * tg1 * 32, T), T, 0, stream>>>(xb, cw1, conv1_b, c1, LSEQ, 998);
        maxpool3<<<nblk((long long)NB * DIM * 332, T), T, 0, stream>>>(c1, p1, 998, 332);
        int tg2 = (330 + 63) / 64;
        wmma_conv<<<nblk((long long)NB * 8 * tg2 * 32, T), T, 0, stream>>>(p1, cw2, conv2_b, c2, 332, 330);
        maxpool3<<<nblk((long long)NB * DIM * 110, T), T, 0, stream>>>(c2, p2, 330, 110);
        int tg3 = (108 + 63) / 64;
        wmma_conv<<<nblk((long long)NB * 8 * tg3 * 32, T), T, 0, stream>>>(p2, cw3, conv3_b, c3, 110, 108);
        pool_all<<<nblk((long long)NB * DIM, T), T, 0, stream>>>(c3, fvec, 108);
    }
    sgemm_small<<<nblk((long long)NB * 128, T), T, 0, stream>>>(fvec, tf_w, tf_b, seq1, NB, 128, 128, 1);

    // ---- fusion + MLP head + softmax ----
    fuse_kernel<<<nblk((long long)NB * 128, T), T, 0, stream>>>(g1, seq1, w1s, gc, NB * 128);
    sgemm_small<<<nblk((long long)NB * 512, T), T, 0, stream>>>(gc, fc1_w, fc1_b, t1, NB, 128, 512, 1);
    sgemm_small<<<nblk((long long)NB * 256, T), T, 0, stream>>>(t1, fc2_w, fc2_b, t2, NB, 512, 256, 1);
    sgemm_small<<<nblk((long long)NB * 2, T), T, 0, stream>>>(t2, out_w, out_b, o2, NB, 256, 2, 1);
    softmax2<<<nblk(NB, T), T, 0, stream>>>(o2, (float*)d_out);

    (void)in_sizes; (void)n_in; (void)out_size; (void)ws_size;
}
